// SplAtconv2d_18159121727497
// MI455X (gfx1250) — matrix operand order repaired and checked by reading, not compiled, not run on hardware
//
#include <hip/hip_runtime.h>
#include <math.h>

// ---------------------------------------------------------------------------
// SplAtConv fused for MI455X (gfx1250, wave32, WMMA).
//
// x[16,4096,256] f32 -> h = relu(x@w_convT+b) [65536,512] -> attention
// recombine -> out[4096,16,256]. HBM floor: 67MB in + 67MB out = 5.8us.
//
// Matrix path: bf16x3 split (xh*wh + xh*wl + xl*wh) on v_wmma_f32_16x16x32_bf16
// (f32-grade accuracy, ~8x the f32-WMMA rate). 64-row M-tile per block and
// in-register B reuse across 2 row-tiles cut B traffic from L2 to ~1GB.
// x tile is pre-packed to bf16 hi/lo dwords in LDS once, so A fragments are
// pure ds_load_b128 (no per-K-step packing VALU). Column pairing (c, c+256)
// inside each wave lets the attention epilogue run register-only.
// ---------------------------------------------------------------------------

typedef __attribute__((ext_vector_type(16))) __bf16 v16bf;
typedef __attribute__((ext_vector_type(8)))  float  v8f;

#define CHN    256
#define CIN    256
#define INTERC 128
#define DOUT   512      // RADIX * CH
#define NPOS   4096
#define BATCH  16
#define XPITCH 132      // dword pitch for packed x tile (bank-spread)

__device__ __forceinline__ unsigned short f32_to_bf16(float f) {
  unsigned int u = __float_as_uint(f);
  u += 0x7FFFu + ((u >> 16) & 1u);            // round-to-nearest-even
  return (unsigned short)(u >> 16);
}
__device__ __forceinline__ float bf16_to_f32(unsigned short h) {
  return __uint_as_float(((unsigned int)h) << 16);
}

union Frag {                                   // 32B: one 16x32 bf16 A/B fragment
  unsigned int u[8];
  uint4 q[2];
  v16bf v;
};

// ---------------------------------------------------------------------------
// K0: split w_conv (f32 [512][256]) into bf16 hi/lo planes (stay hot in L2).
// ---------------------------------------------------------------------------
__global__ void wconv_split_kernel(const float* __restrict__ w,
                                   unsigned short* __restrict__ whi,
                                   unsigned short* __restrict__ wlo) {
  int i = blockIdx.x * blockDim.x + threadIdx.x;
  if (i < DOUT * CIN) {
    float f = w[i];
    unsigned short h = f32_to_bf16(f);
    whi[i] = h;
    wlo[i] = f32_to_bf16(f - bf16_to_f32(h));
  }
}

// ---------------------------------------------------------------------------
// K1: attention weights. The reference's rSoftMax flatten+truncate means
// attens depend ONLY on rows n=0 and n=1024 of each batch:
//   attens[b, l] = sigmoid(atten[b,0,l] - atten[b,1024,l]),  l in [0,512).
// Tiny: 16 blocks x 256 threads, plain f32 VALU for full precision.
// ---------------------------------------------------------------------------
__global__ void atten_kernel(const float* __restrict__ x,
                             const float* __restrict__ w_conv,
                             const float* __restrict__ b_conv,
                             const float* __restrict__ w_fc1,
                             const float* __restrict__ b_fc1,
                             const float* __restrict__ w_fc2,
                             const float* __restrict__ b_fc2,
                             float* __restrict__ attw) {
  __shared__ float Xr[CIN];
  __shared__ float H[DOUT];
  __shared__ float GAP[CHN];
  __shared__ float G[INTERC];
  __shared__ float S[2][DOUT];
  const int b = blockIdx.x;
  const int t = threadIdx.x;   // 0..255

  for (int p = 0; p < 2; ++p) {
    const float* xr = x + (size_t)(b * NPOS + p * 1024) * CIN;
    Xr[t] = xr[t];
    __syncthreads();
    for (int dd = 0; dd < 2; ++dd) {                 // conv row + ReLU
      int d = t + dd * 256;
      float a = b_conv[d];
      const float* wr = w_conv + (size_t)d * CIN;
      for (int c = 0; c < CIN; ++c) a = fmaf(Xr[c], wr[c], a);
      H[d] = fmaxf(a, 0.f);
    }
    __syncthreads();
    GAP[t] = H[t] + H[t + 256];                      // radix-split sum
    __syncthreads();
    if (t < INTERC) {                                // fc1 + ReLU
      float a = b_fc1[t];
      const float* wr = w_fc1 + (size_t)t * CHN;
      for (int c = 0; c < CHN; ++c) a = fmaf(GAP[c], wr[c], a);
      G[t] = fmaxf(a, 0.f);
    }
    __syncthreads();
    for (int dd = 0; dd < 2; ++dd) {                 // fc2
      int d = t + dd * 256;
      float a = b_fc2[d];
      const float* wr = w_fc2 + (size_t)d * INTERC;
      for (int i = 0; i < INTERC; ++i) a = fmaf(G[i], wr[i], a);
      S[p][d] = a;
    }
    __syncthreads();
  }
  for (int dd = 0; dd < 2; ++dd) {
    int l = t + dd * 256;
    attw[b * DOUT + l] = 1.0f / (1.0f + __expf(S[1][l] - S[0][l]));
  }
}

// ---------------------------------------------------------------------------
// K2: fused conv-GEMM + bias + ReLU + attention recombine + [N,B,C] store.
// Block = 256 threads (8 waves), 64-row M-tile. Wave w: colgroup g = w&3
// (cols [g*64, g*64+64) and their +256 partners), rowpair p = w>>2
// (row-tiles p*32 and p*32+16). 16 accum tiles/wave; B fragments reused
// across both row-tiles; attention recombine done in registers.
// ---------------------------------------------------------------------------
__global__ void __launch_bounds__(256)
splat_main_kernel(const float* __restrict__ x,
                  const unsigned short* __restrict__ whi,
                  const unsigned short* __restrict__ wlo,
                  const float* __restrict__ b_conv,
                  const float* __restrict__ attw,
                  float* __restrict__ out) {
  __shared__ unsigned int XhS[64][XPITCH];   // 33 KB packed bf16-hi pairs
  __shared__ unsigned int XlS[64][XPITCH];   // 33 KB packed bf16-lo pairs
  __shared__ float AtS[DOUT];                // attention weights, this batch
  __shared__ float BcS[DOUT];                // conv bias

  const int t    = threadIdx.x;
  const int wv   = t >> 5;              // wave 0..7
  const int lane = t & 31;
  const int g    = wv & 3;              // colgroup: cols g*64.. (+256 pair)
  const int rp   = wv >> 2;             // rowpair: rows rp*32 .. rp*32+31
  const int mrow = lane & 15;           // A-frag row / B-frag col within tile
  const int half = lane >> 4;           // lane half selects K sub-range
  const int m0     = blockIdx.x * 64;   // 64 rows, never crosses a batch
  const int b      = m0 >> 12;          // 4096 rows per batch
  const int n_base = m0 & (NPOS - 1);

  // ---- stage: x tile f32 -> packed bf16 hi/lo dwords in LDS (once) ----
  {
    const float2* src = (const float2*)(x + (size_t)m0 * CIN);
    for (int pi = t; pi < 64 * (CIN / 2); pi += 256) {   // 8192 f32 pairs
      float2 f = src[pi];
      int row = pi >> 7, cp = pi & 127;
      unsigned short h0 = f32_to_bf16(f.x), h1 = f32_to_bf16(f.y);
      unsigned short l0 = f32_to_bf16(f.x - bf16_to_f32(h0));
      unsigned short l1 = f32_to_bf16(f.y - bf16_to_f32(h1));
      XhS[row][cp] = (unsigned int)h0 | ((unsigned int)h1 << 16);
      XlS[row][cp] = (unsigned int)l0 | ((unsigned int)l1 << 16);
    }
    for (int i = t; i < DOUT; i += 256) {
      AtS[i] = attw[b * DOUT + i];
      BcS[i] = b_conv[i];
    }
  }
  __syncthreads();

  v8f acc[16] = {};   // [ (rt*4 + ct)*2 + s ], s: 0 = col, 1 = col+256

  for (int kb = 0; kb < CIN; kb += 32) {
    // A hi/lo fragments for both row-tiles: pure ds_load_b128
    // (ISA 16-bit A layout: lane half selects K+0/K+8; pairs are dwords)
    Frag ah[2], al[2];
#pragma unroll
    for (int rt = 0; rt < 2; ++rt) {
      int row = rp * 32 + rt * 16 + mrow;
      int d0 = (kb >> 1) + half * 4;
      ah[rt].q[0] = *(const uint4*)&XhS[row][d0];
      ah[rt].q[1] = *(const uint4*)&XhS[row][d0 + 8];
      al[rt].q[0] = *(const uint4*)&XlS[row][d0];
      al[rt].q[1] = *(const uint4*)&XlS[row][d0 + 8];
    }
#pragma unroll
    for (int ct = 0; ct < 4; ++ct) {
#pragma unroll
      for (int s = 0; s < 2; ++s) {
        // B fragment: lane holds col; 8 bf16 at K = kb + half*8 and 8 at
        // K = kb + 16 + half*8 -> two aligned B128 loads (L2-hot weights).
        int col = g * 64 + ct * 16 + s * 256 + mrow;
        size_t off = (size_t)col * CIN + kb + half * 8;
        Frag bh, bl;
        bh.q[0] = *(const uint4*)(whi + off);
        bh.q[1] = *(const uint4*)(whi + off + 16);
        bl.q[0] = *(const uint4*)(wlo + off);
        bl.q[1] = *(const uint4*)(wlo + off + 16);
#pragma unroll
        for (int rt = 0; rt < 2; ++rt) {            // B reused for 2 row-tiles
          int ai = (rt * 4 + ct) * 2 + s;
          acc[ai] = __builtin_amdgcn_wmma_f32_16x16x32_bf16(
              false, ah[rt].v, false, bh.v, (short)0, acc[ai], false, false);
          acc[ai] = __builtin_amdgcn_wmma_f32_16x16x32_bf16(
              false, ah[rt].v, false, bl.v, (short)0, acc[ai], false, false);
          acc[ai] = __builtin_amdgcn_wmma_f32_16x16x32_bf16(
              false, al[rt].v, false, bh.v, (short)0, acc[ai], false, false);
        }
      }
    }
  }

  // ---- register-only epilogue: bias + ReLU + attention recombine ----
  // C/D layout: VGPR v -> row (v + 8*half), col = tile_col + mrow.
#pragma unroll
  for (int rt = 0; rt < 2; ++rt) {
#pragma unroll
    for (int ct = 0; ct < 4; ++ct) {
      int c = g * 64 + ct * 16 + mrow;              // c in [0,256)
      float b0 = BcS[c], b1 = BcS[c + 256];
      float a0 = AtS[c], a1 = AtS[c + 256];
      v8f lo = acc[(rt * 4 + ct) * 2 + 0];
      v8f hi = acc[(rt * 4 + ct) * 2 + 1];
      int row0 = n_base + rp * 32 + rt * 16 + 8 * half;
#pragma unroll
      for (int v = 0; v < 8; ++v) {
        float h0 = fmaxf(lo[v] + b0, 0.f);
        float h1 = fmaxf(hi[v] + b1, 0.f);
        out[(size_t)(row0 + v) * (BATCH * CHN) + b * CHN + c] = a0 * h0 + a1 * h1;
      }
    }
  }
}

// ---------------------------------------------------------------------------
extern "C" void kernel_launch(void* const* d_in, const int* in_sizes, int n_in,
                              void* d_out, int out_size, void* d_ws, size_t ws_size,
                              hipStream_t stream) {
  const float* x      = (const float*)d_in[0];
  const float* w_conv = (const float*)d_in[1];
  const float* b_conv = (const float*)d_in[2];
  const float* w_fc1  = (const float*)d_in[3];
  const float* b_fc1  = (const float*)d_in[4];
  const float* w_fc2  = (const float*)d_in[5];
  const float* b_fc2  = (const float*)d_in[6];
  float* out = (float*)d_out;

  char* ws = (char*)d_ws;
  float*          attw = (float*)ws;                         // 16*512 f32  (32 KB)
  unsigned short* whi  = (unsigned short*)(ws + 32768);      // 512*256 u16 (256 KB)
  unsigned short* wlo  = (unsigned short*)(ws + 32768 + 262144);

  wconv_split_kernel<<<dim3(512), dim3(256), 0, stream>>>(w_conv, whi, wlo);
  atten_kernel<<<dim3(16), dim3(256), 0, stream>>>(
      x, w_conv, b_conv, w_fc1, b_fc1, w_fc2, b_fc2, attw);
  splat_main_kernel<<<dim3(NPOS * BATCH / 64), dim3(256), 0, stream>>>(
      x, whi, wlo, b_conv, attw, out);
}
